// HeteroNet_42382737277595
// MI455X (gfx1250) — matrix operand order repaired and checked by reading, not compiled, not run on hardware
//
#include <hip/hip_runtime.h>
#include <hip/hip_bf16.h>
#include <math.h>

typedef __attribute__((ext_vector_type(16))) __bf16 v16bf;
typedef __attribute__((ext_vector_type(8)))  float  v8f;

#define NODES 3840
#define NPG   60
#define BATCH 64
#define GENES 4096
#define HID   512
#define GEDIM 256
#define EDGES 15360
#define TOPK  1000

__device__ __forceinline__ unsigned fkey(float f) {
  unsigned u = __float_as_uint(f);
  return (u & 0x80000000u) ? ~u : (u | 0x80000000u);
}

// ---------------- f32 -> bf16 (optionally transposed) weight staging ----------------
__global__ void cvt_bf16(const float* __restrict__ src, __bf16* __restrict__ dst,
                         int R, int C, int transpose) {
  int i = blockIdx.x * blockDim.x + threadIdx.x;
  if (i >= R * C) return;
  int r = i / C, c = i - r * C;
  __bf16 v = (__bf16)src[i];
  if (transpose) dst[(size_t)c * R + r] = v;
  else           dst[i] = v;
}

// ---------------- generic one-wave-per-tile bf16 WMMA GEMM ----------------
// C[m,n] = act( (sum_k A[m,k]*Wt[n,k] + bias[n]) * scale ),  A f32 [M,K], Wt bf16 [N,K]
__global__ void __launch_bounds__(32) gemm_bf16(
    const float* __restrict__ A, int lda,
    const __bf16* __restrict__ Wt, int K,
    const float* __restrict__ bias,
    float* __restrict__ C, int ldc,
    float scale, int do_relu)
{
  const int lane = threadIdx.x & 31;
  const int half = lane >> 4;
  const int mn   = lane & 15;
  const int m0 = blockIdx.x * 16;
  const int n0 = blockIdx.y * 16;
  const float*  arow = A  + (size_t)(m0 + mn) * lda;
  const __bf16* brow = Wt + (size_t)(n0 + mn) * K;
  v8f acc = {};
  for (int k0 = 0; k0 < K; k0 += 32) {
    __builtin_prefetch(arow + k0 + 32, 0, 3);
    v16bf af, bf;
#pragma unroll
    for (int v = 0; v < 8; ++v) {
      int ka = k0 + 16*(v>>2) + 8*half + 2*(v&3);
      af[2*v]   = (__bf16)arow[ka];
      af[2*v+1] = (__bf16)arow[ka+1];
      int kb = k0 + 16*(v>>2) + 8*half + 2*(v&3);
      bf[2*v]   = brow[kb];
      bf[2*v+1] = brow[kb+1];
    }
    acc = __builtin_amdgcn_wmma_f32_16x16x32_bf16(false, af, false, bf,
                                                  (short)0, acc, false, false);
  }
  const int col = n0 + mn;
  const float bb = bias ? bias[col] : 0.0f;
#pragma unroll
  for (int r = 0; r < 8; ++r) {
    float v = (acc[r] + bb) * scale;
    if (do_relu) v = fmaxf(v, 0.0f);
    C[(size_t)(m0 + 8*half + r) * ldc + col] = v;
  }
}

// ---------------- GNN: node embed + deterministic in-graph edge aggregation ----------------
__global__ void __launch_bounds__(512) gnn_agg(
    const float* __restrict__ x, const int* __restrict__ ei,
    const float* __restrict__ w, const float* __restrict__ b,
    float* __restrict__ agg)
{
  const int g = blockIdx.x;             // one block per 60-node graph
  __shared__ float xs[NPG];
  __shared__ short ssrc[4 * NPG];
  __shared__ int   cstart[NPG + 1];
  __shared__ int   hist[NPG];
  const int t = threadIdx.x;
  if (t < NPG) { xs[t] = x[g * NPG + t]; hist[t] = 0; }
  __syncthreads();
  if (t == 0) {                          // deterministic counting sort by dst
    const int e0 = g * 4 * NPG;
    for (int e = 0; e < 4 * NPG; ++e) hist[ei[EDGES + e0 + e] - g * NPG]++;
    int run = 0;
    for (int d = 0; d < NPG; ++d) { cstart[d] = run; run += hist[d]; hist[d] = cstart[d]; }
    cstart[NPG] = run;
    for (int e = 0; e < 4 * NPG; ++e) {
      int s = ei[e0 + e] - g * NPG;
      int d = ei[EDGES + e0 + e] - g * NPG;
      ssrc[hist[d]++] = (short)s;
    }
  }
  __syncthreads();
  const float wc = w[t], bc = b[t];      // t = channel (512)
  for (int d = 0; d < NPG; ++d) {
    float acc = 0.f;
    for (int j = cstart[d]; j < cstart[d + 1]; ++j)
      acc += fmaxf(xs[ssrc[j]] * wc + bc, 0.f);
    agg[(size_t)(g * NPG + d) * HID + t] = acc;
  }
}

__global__ void __launch_bounds__(512) pool_mean(const float* __restrict__ h2,
                                                 float* __restrict__ pooled) {
  const int b = blockIdx.x, c = threadIdx.x;
  float s = 0.f;
  for (int d = 0; d < NPG; ++d) s += h2[(size_t)(b * NPG + d) * HID + c];
  pooled[(size_t)b * HID + c] = s * (1.0f / NPG);
}

// ---------------- fused per-gene MLP + dot(t_b): logits[b,g] ----------------
// one wave handles 16 genes of one batch row; 64 WMMAs per tile
__global__ void __launch_bounds__(32) sim_logits(
    const float* __restrict__ expv, const float* __restrict__ cw1,
    const float* __restrict__ cb1, const __bf16* __restrict__ cw2t,
    const float* __restrict__ cb2, const float* __restrict__ tmat,
    float* __restrict__ logits)
{
  const int lane = threadIdx.x & 31;
  const int half = lane >> 4;
  const int mn   = lane & 15;
  const int b  = blockIdx.x >> 8;            // 256 tiles per batch row
  const int g0 = (blockIdx.x & 255) << 4;
  const float x = expv[(size_t)b * GENES + g0 + mn];
  v16bf af[4];                               // h1 = relu(x*cw1+cb1), K=128
#pragma unroll
  for (int kc = 0; kc < 4; ++kc) {
#pragma unroll
    for (int v = 0; v < 8; ++v) {
      int k = kc * 32 + 16*(v>>2) + 8*half + 2*(v&3);
      af[kc][2*v]   = (__bf16)fmaxf(x * cw1[k]     + cb1[k],     0.f);
      af[kc][2*v+1] = (__bf16)fmaxf(x * cw1[k + 1] + cb1[k + 1], 0.f);
    }
  }
  float part[8];
#pragma unroll
  for (int r = 0; r < 8; ++r) part[r] = 0.f;
  for (int nt = 0; nt < 16; ++nt) {          // 256 hidden cols in 16 tiles
    const __bf16* brow = cw2t + (size_t)(nt * 16 + mn) * 128;
    v8f acc = {};
#pragma unroll
    for (int kc = 0; kc < 4; ++kc) {
      v16bf bf;
#pragma unroll
      for (int v = 0; v < 8; ++v) {
        int k = kc * 32 + 16*(v>>2) + 8*half + 2*(v&3);
        bf[2*v] = brow[k]; bf[2*v+1] = brow[k + 1];
      }
      acc = __builtin_amdgcn_wmma_f32_16x16x32_bf16(false, af[kc], false, bf,
                                                    (short)0, acc, false, false);
    }
    const int ncol = nt * 16 + mn;
    const float bb = cb2[ncol];
    const float tv = tmat[(size_t)b * GEDIM + ncol];
#pragma unroll
    for (int r = 0; r < 8; ++r)
      part[r] += fmaxf(acc[r] + bb, 0.f) * tv;
  }
#pragma unroll
  for (int m = 1; m < 16; m <<= 1)
#pragma unroll
    for (int r = 0; r < 8; ++r)
      part[r] += __shfl_xor(part[r], m, 32);
  if (mn == 0) {
#pragma unroll
    for (int r = 0; r < 8; ++r)
      logits[(size_t)b * GENES + g0 + half * 8 + r] = part[r];
  }
}

// ---------------- deterministic radix-select top-K mask ----------------
__global__ void __launch_bounds__(256) topk_mask(
    const float* __restrict__ logits, const float* __restrict__ expv,
    float* __restrict__ masked)
{
  const int b = blockIdx.x, tid = threadIdx.x;
  __shared__ int hist[256];
  __shared__ unsigned sprefix, spmask;
  __shared__ int sremain;
  if (tid == 0) { sprefix = 0; spmask = 0; sremain = TOPK; }
  __syncthreads();
  for (int round = 3; round >= 0; --round) {
    hist[tid] = 0;
    __syncthreads();
    const unsigned pref = sprefix, pmask = spmask;
    for (int g = tid; g < GENES; g += 256) {
      unsigned k = fkey(logits[(size_t)b * GENES + g]);
      if ((k & pmask) == pref) atomicAdd(&hist[(k >> (8 * round)) & 255], 1);
    }
    __syncthreads();
    if (tid == 0) {
      int rem = sremain, cum = 0, chosen = 0;
      for (int bin = 255; bin >= 0; --bin) {
        int c = hist[bin];
        if (cum + c >= rem) { chosen = bin; sremain = rem - cum; break; }
        cum += c;
      }
      sprefix = pref | ((unsigned)chosen << (8 * round));
      spmask  = pmask | (0xFFu << (8 * round));
    }
    __syncthreads();
  }
  const unsigned T = sprefix;
  for (int g = tid; g < GENES; g += 256) {
    unsigned k = fkey(logits[(size_t)b * GENES + g]);
    masked[(size_t)b * GENES + g] = (k > T) ? expv[(size_t)b * GENES + g] : 0.f;
  }
  __threadfence_block();
  __syncthreads();
  if (tid == 0) {                            // ties: lowest index first (deterministic)
    int need = sremain;
    for (int g = 0; g < GENES && need > 0; ++g) {
      if (fkey(logits[(size_t)b * GENES + g]) == T) {
        masked[(size_t)b * GENES + g] = expv[(size_t)b * GENES + g];
        --need;
      }
    }
  }
}

// ---------------- final [64,128] @ [128,1] ----------------
__global__ void __launch_bounds__(64) head_out(
    const float* __restrict__ h, const float* __restrict__ w,
    const float* __restrict__ bsc, float* __restrict__ out)
{
  const int b = threadIdx.x;
  float acc = bsc[0];
  for (int i = 0; i < 128; ++i) acc += h[b * 128 + i] * w[i];
  out[b] = acc;
}

extern "C" void kernel_launch(void* const* d_in, const int* in_sizes, int n_in,
                              void* d_out, int out_size, void* d_ws, size_t ws_size,
                              hipStream_t stream) {
  (void)in_sizes; (void)n_in; (void)out_size; (void)ws_size;
  const float* graph_x = (const float*)d_in[0];
  const int*   eidx    = (const int*)d_in[1];
  /* d_in[2] = batch: implicit via row/60 */
  const float* expv    = (const float*)d_in[3];
  const float* drug    = (const float*)d_in[4];
  const float* gw_in = (const float*)d_in[5];  const float* gb_in = (const float*)d_in[6];
  const float* gw_msg= (const float*)d_in[7];  const float* gb_msg= (const float*)d_in[8];
  const float* gw_out= (const float*)d_in[9];  const float* gb_out= (const float*)d_in[10];
  const float* cw1   = (const float*)d_in[11]; const float* cb1   = (const float*)d_in[12];
  const float* cw2   = (const float*)d_in[13]; const float* cb2   = (const float*)d_in[14];
  const float* cw3   = (const float*)d_in[15]; /* cb3 (d_in[16]) adds a per-row const: rank-invariant */
  const float* ew1   = (const float*)d_in[17]; const float* eb1   = (const float*)d_in[18];
  const float* ew2   = (const float*)d_in[19]; const float* eb2   = (const float*)d_in[20];
  const float* ew3   = (const float*)d_in[21]; const float* eb3   = (const float*)d_in[22];
  const float* dw1   = (const float*)d_in[23]; const float* db1   = (const float*)d_in[24];
  const float* dw2   = (const float*)d_in[25]; const float* db2   = (const float*)d_in[26];
  const float* dw3   = (const float*)d_in[27]; const float* db3   = (const float*)d_in[28];
  const float* fw    = (const float*)d_in[29]; const float* fb    = (const float*)d_in[30];
  const float* clw   = (const float*)d_in[31]; const float* clb   = (const float*)d_in[32];
  const float* f2w   = (const float*)d_in[33]; const float* f2b   = (const float*)d_in[34];
  float* out = (float*)d_out;

  char* p = (char*)d_ws;
  auto alloc = [&](size_t bytes) -> void* {
    void* r = (void*)p; p += (bytes + 255) & ~(size_t)255; return r;
  };
  // bf16 weight copies (Wt layout: [N,K] row-major)
  __bf16* gw_msgT = (__bf16*)alloc((size_t)512*512*2);
  __bf16* gw_outT = (__bf16*)alloc((size_t)256*512*2);
  __bf16* cw3B    = (__bf16*)alloc((size_t)256*256*2);   // t = graph_out @ cw3^T -> Wt = cw3
  __bf16* cw2T    = (__bf16*)alloc((size_t)256*128*2);
  __bf16* ew1T    = (__bf16*)alloc((size_t)128*4096*2);
  __bf16* ew2T    = (__bf16*)alloc((size_t)128*128*2);
  __bf16* ew3T    = (__bf16*)alloc((size_t)128*128*2);
  __bf16* dw1T    = (__bf16*)alloc((size_t)128*2048*2);
  __bf16* dw2T    = (__bf16*)alloc((size_t)128*128*2);
  __bf16* dw3T    = (__bf16*)alloc((size_t)128*128*2);
  __bf16* fwT     = (__bf16*)alloc((size_t)128*256*2);
  __bf16* clwT    = (__bf16*)alloc((size_t)128*128*2);
  // f32 activations
  float* agg      = (float*)alloc((size_t)NODES*HID*4);
  float* h2       = (float*)alloc((size_t)NODES*HID*4);
  float* pooled   = (float*)alloc((size_t)BATCH*HID*4);
  float* graphout = (float*)alloc((size_t)BATCH*GEDIM*4);
  float* tmat     = (float*)alloc((size_t)BATCH*GEDIM*4);
  float* logits   = (float*)alloc((size_t)BATCH*GENES*4);
  float* masked   = (float*)alloc((size_t)BATCH*GENES*4);
  float* ch1      = (float*)alloc((size_t)BATCH*128*4);
  float* ch2      = (float*)alloc((size_t)BATCH*128*4);
  float* concatb  = (float*)alloc((size_t)BATCH*256*4);
  float* dh1      = (float*)alloc((size_t)BATCH*128*4);
  float* dh2      = (float*)alloc((size_t)BATCH*128*4);
  float* head1    = (float*)alloc((size_t)BATCH*128*4);
  float* head2    = (float*)alloc((size_t)BATCH*128*4);

  auto cvt = [&](const float* s, __bf16* d, int R, int C, int tr) {
    int n = R * C;
    cvt_bf16<<<(n + 255) / 256, 256, 0, stream>>>(s, d, R, C, tr);
  };
  cvt(gw_msg, gw_msgT, 512, 512, 1);
  cvt(gw_out, gw_outT, 512, 256, 1);
  cvt(cw3,    cw3B,    256, 256, 0);
  cvt(cw2,    cw2T,    128, 256, 1);
  cvt(ew1,    ew1T,   4096, 128, 1);
  cvt(ew2,    ew2T,    128, 128, 1);
  cvt(ew3,    ew3T,    128, 128, 1);
  cvt(dw1,    dw1T,   2048, 128, 1);
  cvt(dw2,    dw2T,    128, 128, 1);
  cvt(dw3,    dw3T,    128, 128, 1);
  cvt(fw,     fwT,     256, 128, 1);
  cvt(clw,    clwT,    128, 128, 1);

  auto gemm = [&](const float* A, int lda, const __bf16* Wt, int K, const float* bias,
                  float* C, int ldc, int M, int N, float scale, int relu) {
    dim3 grid(M / 16, N / 16);
    gemm_bf16<<<grid, 32, 0, stream>>>(A, lda, Wt, K, bias, C, ldc, scale, relu);
  };

  // ---- GNN ----
  gnn_agg<<<BATCH, 512, 0, stream>>>(graph_x, eidx, gw_in, gb_in, agg);
  gemm(agg, HID, gw_msgT, HID, gb_msg, h2, HID, NODES, HID, 1.0f, 1);     // relu
  pool_mean<<<BATCH, 512, 0, stream>>>(h2, pooled);
  gemm(pooled, HID, gw_outT, HID, gb_out, graphout, GEDIM, BATCH, GEDIM, 1.0f, 0);
  gemm(graphout, GEDIM, cw3B, GEDIM, nullptr, tmat, GEDIM, BATCH, GEDIM, 1.0f, 0);

  // ---- fused per-gene MLP + dot ----
  sim_logits<<<BATCH * (GENES / 16), 32, 0, stream>>>(expv, cw1, cb1, cw2T, cb2, tmat, logits);
  topk_mask<<<BATCH, 256, 0, stream>>>(logits, expv, masked);

  // ---- encoders ----
  gemm(masked, GENES, ew1T, GENES, eb1, ch1, 128, BATCH, 128, 1.0f, 1);
  gemm(ch1, 128, ew2T, 128, eb2, ch2, 128, BATCH, 128, 1.0f, 1);
  gemm(ch2, 128, ew3T, 128, eb3, concatb, 256, BATCH, 128, 1.0f, 0);        // cols 0..127
  gemm(drug, 2048, dw1T, 2048, db1, dh1, 128, BATCH, 128, 1.0f, 1);
  gemm(dh1, 128, dw2T, 128, db2, dh2, 128, BATCH, 128, 1.0f, 1);
  gemm(dh2, 128, dw3T, 128, db3, concatb + 128, 256, BATCH, 128, 1.0f, 0);  // cols 128..255

  // ---- predictor head ----
  const float invs = 1.0f / sqrtf(1.0f + 1e-5f);
  gemm(concatb, 256, fwT, 256, fb, head1, 128, BATCH, 128, 1.0f, 1);
  gemm(head1, 128, clwT, 128, clb, head2, 128, BATCH, 128, invs, 1);        // (x@clw+clb)*invs, relu
  head_out<<<1, 64, 0, stream>>>(head2, f2w, f2b, out);
}
